// RNN_Clusterer_34308198760902
// MI455X (gfx1250) — compile-verified, matrix operand order checked by reading, not run on hardware
//
#include <hip/hip_runtime.h>
#include <hip/hip_bf16.h>

// ---------------------------------------------------------------------------
// RNN speaker-clusterer for MI455X (gfx1250, wave32, WMMA).
// Persistent WGs: 32 blocks x 256 threads (8 waves). Each block owns 4
// batches (M = 16 state rows) and iterates all 500 chunks sequentially with
// cluster state in LDS. Per step: tiny scoring dot-products + log-softmax
// (VALU), then two (16x256)x(256x768) GEMMs on v_wmma_f32_16x16x32_bf16
// (fp32 accumulate), gates, masked cluster update.
// Weights are pre-converted to bf16 and pre-swizzled into per-lane WMMA
// A-tile order so the hot loop streams them with coalesced b128 loads
// straight out of the 192MB L2 (weights are only 768KB packed).
// ---------------------------------------------------------------------------

typedef __attribute__((ext_vector_type(16))) __bf16 v16bf;
typedef __attribute__((ext_vector_type(8)))  float  v8f;

constexpr int Nb = 128;   // batch
constexpr int Cn = 500;   // chunks (sequential)
constexpr int Sm = 4;     // max speakers
constexpr int Dm = 256;   // hidden units
constexpr int G3 = 3 * Dm;           // 768 gate features
constexpr int FT = G3 / 16;          // 48 feature tiles
constexpr int KT = Dm / 32;          // 8 k-chunks
constexpr int WGB = 4;               // batches per workgroup
constexpr int MROWS = WGB * Sm;      // 16 state rows per WG
constexpr int PACKED_W_ELEMS = 2 * FT * KT * 32 * 16;  // 393216 bf16

// ISA 16-bit A/B tile lane layout (16x32 bf16): lane l (l<16) holds row l,
// slots 0..7 -> K 0..7, slots 8..15 -> K 16..23; lanes 16..31 hold row l-16
// with K offset +8.
__device__ __forceinline__ int kmap(int j, int l) {
  int k = (j < 8) ? j : (j + 8);
  return (l >= 16) ? (k + 8) : k;
}

// ---------------------------------------------------------------------------
// Kernel 1: pack W_ih / W_hh (768x256 fp32, row = out feature) into bf16
// WMMA A-tiles: Wp[mat][ftile][ktile][lane][slot]
// ---------------------------------------------------------------------------
__global__ void pack_weights_kernel(const float* __restrict__ W_ih,
                                    const float* __restrict__ W_hh,
                                    __bf16* __restrict__ Wp) {
  int idx = blockIdx.x * blockDim.x + threadIdx.x;
  if (idx >= PACKED_W_ELEMS) return;
  int j   = idx & 15;
  int l   = (idx >> 4) & 31;
  int kt  = (idx >> 9) & 7;
  int ft  = (idx >> 12) % FT;
  int mat = idx / (FT * KT * 32 * 16);
  int row = ft * 16 + (l & 15);
  int k   = kt * 32 + kmap(j, l);
  const float* W = mat ? W_hh : W_ih;
  Wp[idx] = (__bf16)W[row * Dm + k];
}

// ---------------------------------------------------------------------------
// Kernel 2: persistent per-4-batch recurrence.
// ---------------------------------------------------------------------------
__global__ __launch_bounds__(256, 1) void rnn_cluster_kernel(
    const float* __restrict__ spk_emb,   // (N,C,S,D)
    const int*   __restrict__ spk_nums,  // (N,C)
    const int*   __restrict__ label,     // (N,C,S)
    const float* __restrict__ b_ih,      // (768)
    const float* __restrict__ b_hh,      // (768)
    const float* __restrict__ init_h,    // (D)
    const __bf16* __restrict__ Wp,       // packed weights
    float* __restrict__ losses)          // (32) per-WG partial loss
{
  __shared__ __align__(16) float Hs[MROWS][Dm + 4];   // cluster state (fp32)
  __shared__ __align__(16) float Es[MROWS][Dm + 4];   // this step's embeddings
  __shared__ __align__(64) __bf16 Hb[KT][32][16];     // h packed as B tiles
  __shared__ __align__(64) __bf16 Xb[KT][32][16];     // ordered x packed as B
  __shared__ float dotp[64][4];
  __shared__ float picked16[MROWS];
  __shared__ int   labS[MROWS];
  __shared__ int   updS[MROWS];
  __shared__ int   nspkS[WGB];
  __shared__ float biasI[G3], biasH[G3];
  __shared__ float lossAcc;

  const int tid  = threadIdx.x;
  const int lane = tid & 31;
  const int wave = tid >> 5;
  const int b0   = blockIdx.x * WGB;

  // one-time init
  for (int f = tid; f < G3; f += 256) { biasI[f] = b_ih[f]; biasH[f] = b_hh[f]; }
  for (int idx = tid; idx < MROWS * Dm; idx += 256) {
    int m = idx >> 8, d = idx & (Dm - 1);
    Hs[m][d] = init_h[d];
  }
  if (tid == 0) lossAcc = 0.0f;

  // feature tiles owned by this wave: r,z,n triples share (f mod 256) range
  const int ft0 = wave * 2;
  const int ftile[6] = { ft0, ft0 + 1, 16 + ft0, 17 + ft0, 32 + ft0, 33 + ft0 };
  const v8f zero8 = {0.f, 0.f, 0.f, 0.f, 0.f, 0.f, 0.f, 0.f};

  __syncthreads();

  for (int c = 0; c < Cn; ++c) {
    // ---- stage labels / counts / embeddings (pre-B1) -------------------
    if (tid < MROWS) {
      int bl = tid >> 2, i = tid & 3;
      labS[tid] = label[(((size_t)(b0 + bl)) * Cn + c) * Sm + i];
    }
    if (tid < WGB) nspkS[tid] = spk_nums[((size_t)(b0 + tid)) * Cn + c];
    for (int idx = tid; idx < MROWS * (Dm / 4); idx += 256) {
      int m = idx >> 6, q = idx & 63;
      const float4* src = (const float4*)(spk_emb +
          ((((size_t)(b0 + (m >> 2))) * Cn + c) * Sm + (m & 3)) * Dm);
      *(float4*)&Es[m][q * 4] = src[q];
    }
    __syncthreads();  // B1: Es/labS/nspkS ready; all prior-step Hs updates done

    // ---- update mask ----------------------------------------------------
    if (tid < MROWS) {
      int bl = tid >> 2, j = tid & 3, u = 0;
      for (int i = 0; i < Sm; ++i)
        u |= (labS[bl * 4 + i] == j) && (i < nspkS[bl]);
      updS[tid] = u;
    }
    // ---- scoring: 64 dot products (emb_i . H_j), 4 partials each -------
    {
      int dot = tid >> 2, part = tid & 3;
      int bl = dot >> 4, i = (dot >> 2) & 3, j = dot & 3;
      const float* e = Es[bl * 4 + i];
      const float* h = Hs[bl * 4 + j];
      float s = 0.f;
      int d0 = part * 64;
      for (int d = 0; d < 64; ++d) s += e[d0 + d] * h[d0 + d];
      dotp[dot][part] = s;
    }
    // ---- pack Hb (pre-update h) and Xb (label-gathered x) as B tiles ---
    for (int idx = tid; idx < KT * 32 * 16; idx += 256) {
      int kt = idx >> 9, l = (idx >> 4) & 31, j = idx & 15;
      int m = l & 15;
      int k = kt * 32 + kmap(j, l);
      Hb[kt][l][j] = (__bf16)Hs[m][k];
      int src = (m & ~3) | labS[m];      // emb row = (batch, lab[b,i])
      Xb[kt][l][j] = (__bf16)Es[src][k];
    }
    __syncthreads();  // B1b: dotp ready

    // ---- log-softmax + picked loss -------------------------------------
    if (tid < MROWS) {
      int bl = tid >> 2, i = tid & 3;
      float l4[4];
      for (int j = 0; j < 4; ++j) {
        int dot = bl * 16 + i * 4 + j;
        l4[j] = dotp[dot][0] + dotp[dot][1] + dotp[dot][2] + dotp[dot][3];
      }
      float mx = fmaxf(fmaxf(l4[0], l4[1]), fmaxf(l4[2], l4[3]));
      float lse = logf(__expf(l4[0] - mx) + __expf(l4[1] - mx) +
                       __expf(l4[2] - mx) + __expf(l4[3] - mx)) + mx;
      float pick = l4[labS[tid]] - lse;
      picked16[tid] = (i < nspkS[bl]) ? -pick : 0.0f;
    }
    __syncthreads();  // B2: Hb/Xb/picked16 ready

    if (tid == 0) {
      float s = 0.f;
      for (int m = 0; m < MROWS; ++m) s += picked16[m];
      lossAcc += s;
    }

    // ---- GEMMs: gi = x @ W_ih^T, gh = h @ W_hh^T on WMMA bf16 ----------
    v8f acc_i[6], acc_h[6];
#pragma unroll
    for (int t = 0; t < 6; ++t) { acc_i[t] = zero8; acc_h[t] = zero8; }

    for (int kt = 0; kt < KT; ++kt) {
      v16bf bx = *(const v16bf*)&Xb[kt][lane][0];
      v16bf bh = *(const v16bf*)&Hb[kt][lane][0];
      if (kt + 1 < KT)
        __builtin_prefetch(Wp + ((size_t)(ftile[0] * KT + kt + 1) * 512) + lane * 16, 0, 0);
#pragma unroll
      for (int t = 0; t < 6; ++t) {
        const __bf16* pi = Wp + ((size_t)(ftile[t] * KT + kt) * 512) + lane * 16;
        const __bf16* ph = Wp + ((size_t)((FT + ftile[t]) * KT + kt) * 512) + lane * 16;
        v16bf ai = *(const v16bf*)pi;
        v16bf ah = *(const v16bf*)ph;
        acc_i[t] = __builtin_amdgcn_wmma_f32_16x16x32_bf16(
            false, ai, false, bx, (short)0, acc_i[t], false, false);
        acc_h[t] = __builtin_amdgcn_wmma_f32_16x16x32_bf16(
            false, ah, false, bh, (short)0, acc_h[t], false, false);
      }
    }

    // ---- gates + masked cluster update ---------------------------------
    // D-tile layout: element e of lane l -> feature row e + (l>=16 ? 8:0),
    // column = state row (l & 15).
    const int m    = lane & 15;
    const int rofs = (lane >= 16) ? 8 : 0;
    const int u    = updS[m];
#pragma unroll
    for (int t = 0; t < 2; ++t) {
      int fbase = (ft0 + t) * 16;
#pragma unroll
      for (int e = 0; e < 8; ++e) {
        int f = fbase + e + rofs;                   // 0..255 within gate
        float gr = (acc_i[t][e]     + biasI[f])       + (acc_h[t][e]     + biasH[f]);
        float gz = (acc_i[2 + t][e] + biasI[256 + f]) + (acc_h[2 + t][e] + biasH[256 + f]);
        float in = acc_i[4 + t][e] + biasI[512 + f];
        float hn = acc_h[4 + t][e] + biasH[512 + f];
        float r  = 1.0f / (1.0f + __expf(-gr));
        float z  = 1.0f / (1.0f + __expf(-gz));
        float nn = tanhf(in + r * hn);
        float ho = Hs[m][f];
        float hN = (1.0f - z) * nn + z * ho;
        if (u) Hs[m][f] = hN;
      }
    }
    // next iteration's B1 barrier orders these Hs writes vs. readers
  }

  __syncthreads();
  if (tid == 0) losses[blockIdx.x] = lossAcc;
}

// ---------------------------------------------------------------------------
// Kernel 3: deterministic finalize: loss = sum(partials) / sum(spk_nums)
// ---------------------------------------------------------------------------
__global__ void finalize_kernel(const int* __restrict__ spk_nums,
                                const float* __restrict__ losses,
                                float* __restrict__ out) {
  __shared__ int sred[256];
  int tid = threadIdx.x;
  int s = 0;
  for (int i = tid; i < Nb * Cn; i += 256) s += spk_nums[i];
  sred[tid] = s;
  __syncthreads();
  for (int off = 128; off > 0; off >>= 1) {
    if (tid < off) sred[tid] += sred[tid + off];
    __syncthreads();
  }
  if (tid == 0) {
    float L = 0.f;
    for (int i = 0; i < Nb / WGB; ++i) L += losses[i];
    out[0] = L / (float)sred[0];
  }
}

// ---------------------------------------------------------------------------
extern "C" void kernel_launch(void* const* d_in, const int* in_sizes, int n_in,
                              void* d_out, int out_size, void* d_ws, size_t ws_size,
                              hipStream_t stream) {
  const float* spk_emb  = (const float*)d_in[0];
  const int*   spk_nums = (const int*)  d_in[1];
  const int*   label    = (const int*)  d_in[2];
  const float* W_ih     = (const float*)d_in[3];
  const float* W_hh     = (const float*)d_in[4];
  const float* b_ih     = (const float*)d_in[5];
  const float* b_hh     = (const float*)d_in[6];
  const float* init_h   = (const float*)d_in[7];

  __bf16* Wp     = (__bf16*)d_ws;
  float*  losses = (float*)((char*)d_ws + PACKED_W_ELEMS * sizeof(__bf16));

  int packBlocks = (PACKED_W_ELEMS + 255) / 256;
  pack_weights_kernel<<<packBlocks, 256, 0, stream>>>(W_ih, W_hh, Wp);
  rnn_cluster_kernel<<<Nb / WGB, 256, 0, stream>>>(
      spk_emb, spk_nums, label, b_ih, b_hh, init_h, Wp, losses);
  finalize_kernel<<<1, 256, 0, stream>>>(spk_nums, losses, (float*)d_out);
}
